// LoadExperts_54408645706212
// MI455X (gfx1250) — matrix operand order repaired and checked by reading, not compiled, not run on hardware
//
#include <hip/hip_runtime.h>
#include <stdint.h>

// Problem constants: T = B*S = 1024 tokens, H = 1024, I2 = 2048, I = 1024, E = 32.
#define T_TOK  1024
#define H_DIM  1024
#define I_DIM  1024
#define I2_DIM 2048
#define NEXP   32
#define ALPHA_C 1.702f
#define LIMIT_C 7.0f

// Tiling: 256 threads = 8 wave32 waves (2 M x 4 N). Block tile M=32 x N=128,
// each wave owns TWO 16-wide N-tiles -> A fragment reused across 2x WMMAs.
// K staged through LDS in steps of 64 (2 WMMA k-chunks per barrier).
#define TM 32
#define TN 128
#define TK 64

typedef __attribute__((ext_vector_type(16))) __bf16 v16bf;
typedef __attribute__((ext_vector_type(8)))  __bf16 v8bf;
typedef __attribute__((ext_vector_type(4)))  __bf16 v4bf;
typedef __attribute__((ext_vector_type(8)))  float  v8f;

// Native f32->bf16 (RNE) - backend selects v_cvt_pk_bf16_f32 packed forms.
static __device__ __forceinline__ __bf16 f2bf(float f) { return (__bf16)f; }

static __device__ __forceinline__ v16bf cat8(v8bf lo, v8bf hi) {
  return __builtin_shufflevector(lo, hi, 0,1,2,3,4,5,6,7,8,9,10,11,12,13,14,15);
}

// out[t][h] = sum_e rw[t][e] * down_bias[e][h]
__global__ __launch_bounds__(256) void moe_init_out(
    const float* __restrict__ rw, const float* __restrict__ bd,
    float* __restrict__ out)
{
  uint32_t idx = blockIdx.x * 256u + threadIdx.x;   // over T*H = 1M
  uint32_t t = idx >> 10;
  uint32_t h = idx & (H_DIM - 1);
  float s = 0.f;
  #pragma unroll
  for (uint32_t e = 0; e < NEXP; ++e)
    s += rw[t * NEXP + e] * bd[e * H_DIM + h];
  out[idx] = s;
}

// act[t][i] = glu_act( (X @ Wgu)[t, 2i/2i+1] + bias )  for one expert, bf16 out
__global__ __launch_bounds__(256) void moe_gateup_act(
    const float* __restrict__ x,      // [T][H]
    const float* __restrict__ wgu,    // [H][I2]  (this expert's slice)
    const float* __restrict__ bgu,    // [I2]
    unsigned short* __restrict__ act) // [T][I]   (bf16 bits)
{
  __shared__ __align__(32) __bf16 lsA[TM][TK];       // 4 KB  [token][k]
  __shared__ __align__(32) __bf16 lsB[2 * TN][TK];   // 32 KB [gup col][k] (K-transposed)

  const uint32_t t0 = blockIdx.x * TM;
  const uint32_t i0 = blockIdx.y * TN;     // act-column base (128 wide)
  const uint32_t f0 = 2 * i0;              // gate_up column base (256 wide)
  const uint32_t tid  = threadIdx.x;
  const uint32_t lane = tid & 31;
  const uint32_t wave = tid >> 5;
  const uint32_t wm = wave >> 2;           // 0..1  (M sub-tile)
  const uint32_t wn = wave & 3;            // 0..3  (N sub-tile)
  const uint32_t l15 = lane & 15;
  const uint32_t lh  = lane >> 4;

  // WMMA fragment addressing used here (contiguous k order, same for A and B):
  // A 16x32 bf16: lanes0-15 row=l15 K=0..15; lanes16-31 K=16..31
  const uint32_t arow = 16 * wm + l15;
  const uint32_t ak0  = lh ? 16u : 0u;
  // B 32x16 bf16: lanes0-15 col=l15 K=0..15; lanes16-31 K=16..31
  const uint32_t gcol0 = 2 * (16 * wn + l15);   // even gup col (gate), tile 0
  const uint32_t gcol1 = gcol0 + 128;           // tile 1 (+64 act cols)
  const uint32_t bk0   = lh ? 16u : 0u;

  // Hoisted 32-bit global offsets (uniform base ptr + u32 offset -> saddr form)
  uint32_t aoff = (t0 + (tid >> 4)) * H_DIM + ((tid & 15u) << 2);      // A stage, j stride 16*H
  uint32_t boff = (tid >> 6) * I2_DIM + f0 + ((tid & 63u) << 2);       // B stage, j stride 4*I2

  v8f accg0 = {0.f,0.f,0.f,0.f,0.f,0.f,0.f,0.f};
  v8f accu0 = accg0, accg1 = accg0, accu1 = accg0;

  #pragma unroll 2
  for (uint32_t k0 = 0; k0 < H_DIM; k0 += TK, aoff += TK, boff += TK * I2_DIM) {
    // Stage A tile: 32 x 64 f32 -> bf16, packed 4-wide LDS stores
    #pragma unroll
    for (uint32_t j = 0; j < 2; ++j) {
      uint32_t flat = tid + j * 256;
      uint32_t r = flat >> 4;
      uint32_t c = (flat & 15u) << 2;
      float4 v = *(const float4*)(x + aoff + j * (16 * H_DIM));
      v4bf b = { f2bf(v.x), f2bf(v.y), f2bf(v.z), f2bf(v.w) };
      *(v4bf*)(&lsA[r][c]) = b;
    }
    // Stage B tile: 64 k-rows x 256 gup-cols, stored K-transposed
    #pragma unroll 4
    for (uint32_t j = 0; j < 16; ++j) {
      uint32_t flat = tid + j * 256;
      uint32_t r = flat >> 6;                 // k row 0..63
      uint32_t c = (flat & 63u) << 2;         // col 0..252
      float4 v = *(const float4*)(wgu + boff + j * (4 * I2_DIM));
      lsB[c + 0][r] = f2bf(v.x); lsB[c + 1][r] = f2bf(v.y);
      lsB[c + 2][r] = f2bf(v.z); lsB[c + 3][r] = f2bf(v.w);
    }
    if (k0 + TK < H_DIM) {
      __builtin_prefetch(x + aoff + TK, 0, 0);
      __builtin_prefetch(wgu + boff + TK * I2_DIM, 0, 0);
    }
    __syncthreads();

    #pragma unroll
    for (uint32_t kk = 0; kk < TK; kk += 32) {
      v16bf a  = *(const v16bf*)(&lsA[arow][kk + ak0]);
      v16bf bg0 = *(const v16bf*)(&lsB[gcol0    ][kk + bk0]);
      v16bf bu0 = *(const v16bf*)(&lsB[gcol0 + 1][kk + bk0]);
      v16bf bg1 = *(const v16bf*)(&lsB[gcol1    ][kk + bk0]);
      v16bf bu1 = *(const v16bf*)(&lsB[gcol1 + 1][kk + bk0]);
      accg0 = __builtin_amdgcn_wmma_f32_16x16x32_bf16(false, a, false, bg0,
                                                      (short)0, accg0, false, false);
      accu0 = __builtin_amdgcn_wmma_f32_16x16x32_bf16(false, a, false, bu0,
                                                      (short)0, accu0, false, false);
      accg1 = __builtin_amdgcn_wmma_f32_16x16x32_bf16(false, a, false, bg1,
                                                      (short)0, accg1, false, false);
      accu1 = __builtin_amdgcn_wmma_f32_16x16x32_bf16(false, a, false, bu1,
                                                      (short)0, accu1, false, false);
    }
    __syncthreads();
  }

  // Epilogue: bias + clamp + GLU, write bf16 activation.
  // C layout: VGPR r -> M = r + 8*lh, N = l15.
  #pragma unroll
  for (uint32_t p = 0; p < 2; ++p) {
    const v8f accg = p ? accg1 : accg0;
    const v8f accu = p ? accu1 : accu0;
    const uint32_t icol = i0 + 64 * p + 16 * wn + l15;
    const float bg = bgu[2 * icol];
    const float bu = bgu[2 * icol + 1];
    #pragma unroll
    for (uint32_t r = 0; r < 8; ++r) {
      uint32_t t = t0 + 16 * wm + r + 8 * lh;
      float g = accg[r] + bg;
      g = fminf(g, LIMIT_C);
      float u = accu[r] + bu;
      u = fminf(fmaxf(u, -LIMIT_C), LIMIT_C);
      float glu = g / (1.0f + __expf(-ALPHA_C * g));
      float o = (u + 1.0f) * glu;
      act[t * I_DIM + icol] = __builtin_bit_cast(unsigned short, f2bf(o));
    }
  }
}

// out[t][h] += rw[t][e] * (act @ Wd[e])[t][h]   for one expert
__global__ __launch_bounds__(256) void moe_down_acc(
    const unsigned short* __restrict__ act, // [T][I] bf16 bits
    const float* __restrict__ wd,           // [I][H] (this expert's slice)
    const float* __restrict__ rw,           // [T][E]
    const int e,
    float* __restrict__ out)                // [T][H]
{
  __shared__ __align__(32) __bf16 lsA[TM][TK];  // 4 KB
  __shared__ __align__(32) __bf16 lsB[TN][TK];  // 16 KB [h col][k]

  const uint32_t t0 = blockIdx.x * TM;
  const uint32_t h0 = blockIdx.y * TN;
  const uint32_t tid  = threadIdx.x;
  const uint32_t lane = tid & 31;
  const uint32_t wave = tid >> 5;
  const uint32_t wm = wave >> 2;
  const uint32_t wn = wave & 3;
  const uint32_t l15 = lane & 15;
  const uint32_t lh  = lane >> 4;

  const uint32_t arow  = 16 * wm + l15;
  const uint32_t ak0   = lh ? 16u : 0u;
  const uint32_t bcol0 = 16 * wn + l15;
  const uint32_t bcol1 = bcol0 + 64;
  const uint32_t bk0   = lh ? 16u : 0u;

  uint32_t aoff = (t0 + (tid >> 3)) * I_DIM + ((tid & 7u) << 3);   // A stage (bf16 elems)
  uint32_t boff = (tid >> 5) * H_DIM + h0 + ((tid & 31u) << 2);    // B stage, j stride 8*H

  v8f acc0 = {0.f,0.f,0.f,0.f,0.f,0.f,0.f,0.f};
  v8f acc1 = acc0;

  #pragma unroll 2
  for (uint32_t k0 = 0; k0 < I_DIM; k0 += TK, aoff += TK, boff += TK * H_DIM) {
    // Stage A: 32 x 64 bf16, raw copy (one 16B vector per thread)
    {
      uint32_t r = tid >> 3;
      uint32_t c = (tid & 7u) << 3;
      uint4 v = *(const uint4*)(act + aoff);
      *(uint4*)(&lsA[r][c]) = v;
    }
    // Stage B: 64 k-rows x 128 h-cols f32 -> bf16, K-transposed
    #pragma unroll 4
    for (uint32_t j = 0; j < 8; ++j) {
      uint32_t flat = tid + j * 256;
      uint32_t r = flat >> 5;
      uint32_t c = (flat & 31u) << 2;
      float4 v = *(const float4*)(wd + boff + j * (8 * H_DIM));
      lsB[c + 0][r] = f2bf(v.x); lsB[c + 1][r] = f2bf(v.y);
      lsB[c + 2][r] = f2bf(v.z); lsB[c + 3][r] = f2bf(v.w);
    }
    if (k0 + TK < I_DIM) {
      __builtin_prefetch(act + aoff + TK, 0, 0);
      __builtin_prefetch(wd + boff + TK * H_DIM, 0, 0);
    }
    __syncthreads();

    #pragma unroll
    for (uint32_t kk = 0; kk < TK; kk += 32) {
      v16bf a  = *(const v16bf*)(&lsA[arow][kk + ak0]);
      v16bf b0 = *(const v16bf*)(&lsB[bcol0][kk + bk0]);
      v16bf b1 = *(const v16bf*)(&lsB[bcol1][kk + bk0]);
      acc0 = __builtin_amdgcn_wmma_f32_16x16x32_bf16(false, a, false, b0,
                                                     (short)0, acc0, false, false);
      acc1 = __builtin_amdgcn_wmma_f32_16x16x32_bf16(false, a, false, b1,
                                                     (short)0, acc1, false, false);
    }
    __syncthreads();
  }

  #pragma unroll
  for (uint32_t p = 0; p < 2; ++p) {
    const v8f acc = p ? acc1 : acc0;
    const uint32_t h = h0 + 64 * p + 16 * wn + l15;
    #pragma unroll
    for (uint32_t r = 0; r < 8; ++r) {
      uint32_t t = t0 + 16 * wm + r + 8 * lh;
      float w = rw[t * NEXP + (uint32_t)e];
      out[t * H_DIM + h] += w * acc[r];  // serialized across experts
    }
  }
}

extern "C" void kernel_launch(void* const* d_in, const int* in_sizes, int n_in,
                              void* d_out, int out_size, void* d_ws, size_t ws_size,
                              hipStream_t stream) {
  (void)in_sizes; (void)n_in; (void)out_size; (void)ws_size;
  const float* x   = (const float*)d_in[0];   // hidden_states [4,256,1024]
  const float* rw  = (const float*)d_in[1];   // routing_weights [1024,32]
  // d_in[2] router_indices: unused by the dense reference path
  const float* wgu = (const float*)d_in[3];   // gate_up_proj [32,1024,2048]
  const float* bgu = (const float*)d_in[4];   // gate_up_proj_bias [32,2048]
  const float* wd  = (const float*)d_in[5];   // down_proj [32,1024,1024]
  const float* bd  = (const float*)d_in[6];   // down_proj_bias [32,1024]
  float* out = (float*)d_out;                 // [4,256,1024] f32
  unsigned short* act = (unsigned short*)d_ws; // per-expert [1024][1024] bf16 = 2 MB

  moe_init_out<<<(T_TOK * H_DIM) / 256, 256, 0, stream>>>(rw, bd, out);

  dim3 gA(T_TOK / TM, I_DIM / TN);   // 32 x 8 blocks
  dim3 gB(T_TOK / TM, H_DIM / TN);   // 32 x 8 blocks
  for (int e = 0; e < NEXP; ++e) {
    moe_gateup_act<<<gA, 256, 0, stream>>>(
        x, wgu + (size_t)e * H_DIM * I2_DIM, bgu + (size_t)e * I2_DIM, act);
    moe_down_acc<<<gB, 256, 0, stream>>>(
        act, wd + (size_t)e * I_DIM * H_DIM, rw, e, out);
  }
}
